// BasicBlock_37546604102378
// MI455X (gfx1250) — matrix operand order repaired and checked by reading, not compiled, not run on hardware
//
#include <hip/hip_runtime.h>
#include <hip/hip_bf16.h>
#include <stdint.h>

typedef __attribute__((ext_vector_type(16))) __bf16 v16bf;
typedef __attribute__((ext_vector_type(8)))  __bf16 v8bf;
typedef __attribute__((ext_vector_type(8)))  float  v8f;

#define CC 128     // channels
#define KK 9       // taps
#define NT 128     // positions per workgroup tile
#define RS 136     // padded LDS row stride in bf16 elements (272B -> bank rotation)

__device__ __forceinline__ unsigned short f2bf(float f) {
  unsigned u = __float_as_uint(f);
  u += 0x7fffu + ((u >> 16) & 1u);          // round-to-nearest-even
  return (unsigned short)(u >> 16);
}

// A-matrix 16x32 bf16 fragment: per lane, elements 0-7 at p+0..7, 8-15 at p+16..23
__device__ __forceinline__ v16bf load_a_frag(const unsigned short* p) {
  v8bf lo = *(const v8bf*)p;
  v8bf hi = *(const v8bf*)(p + 16);
  v16bf r;
#pragma unroll
  for (int i = 0; i < 8; ++i) { r[i] = lo[i]; r[i + 8] = hi[i]; }
  return r;
}

// B-matrix 32x16 bf16 fragment: per lane, elements 0-7 at p+0..7, 8-15 at p+16..23
__device__ __forceinline__ v16bf load_b_frag(const unsigned short* p) {
  v8bf lo = *(const v8bf*)p;
  v8bf hi = *(const v8bf*)(p + 16);
  v16bf r;
#pragma unroll
  for (int i = 0; i < 8; ++i) { r[i] = lo[i]; r[i + 8] = hi[i]; }
  return r;
}

// ---------------- main WMMA conv kernel -----------------------------------
// D[128, NT] = sum_t sum_c W[o, t*128+c] * (g[t,n] * XP[c, n+t-4])
__global__ __launch_bounds__(256) void conv_wmma_kernel(
    const unsigned short* __restrict__ XP,  // [B][N][128] bf16, curve order
    const unsigned short* __restrict__ Wb,  // [128][1152]  bf16, k' = t*128+c
    const float* __restrict__ G,            // [B][9][N]
    float* __restrict__ Y,                  // [B][N][128] f32, curve order
    int N)
{
  __shared__ unsigned short sXP[(NT + 8) * RS];  // input tile + halo   (~37 KB)
  __shared__ unsigned short sB[NT * RS];         // per-tap gated operand (~35 KB)
  __shared__ float sG[KK * NT];                  // gaussians           (~4.5 KB)

  const int b   = blockIdx.y;
  const int n0  = blockIdx.x * NT;
  const int tid = threadIdx.x;

  { // cooperative tile load (dword granularity), zero halo outside [0,N)
    const size_t base = (size_t)b * N * CC;
    for (int i = tid; i < (NT + 8) * (CC / 2); i += 256) {
      int p = i >> 6, d = i & 63;
      int n = n0 - 4 + p;
      unsigned v = 0u;
      if (n >= 0 && n < N)
        v = *(const unsigned*)(XP + base + (size_t)n * CC + d * 2);
      *(unsigned*)(sXP + p * RS + d * 2) = v;
    }
    for (int i = tid; i < KK * NT; i += 256) {
      int t = i / NT, j = i % NT;
      sG[i] = G[((size_t)b * KK + t) * N + (n0 + j)];
    }
  }
  __syncthreads();

  const int lane = tid & 31;
  const int wave = tid >> 5;
  const int col  = lane & 15;
  const int hi   = lane >> 4;

  v8f acc[8];
#pragma unroll
  for (int nt = 0; nt < 8; ++nt) acc[nt] = (v8f){};

  const unsigned short* Abase = Wb + (size_t)(wave * 16 + col) * (CC * KK) + hi * 8;

  for (int t = 0; t < KK; ++t) {
    // build sB[n][c] = g[t][n] * sXP[n+t][c]  via packed bf16 multiply
    for (int i = tid; i < NT * 16; i += 256) {
      int n = i >> 4, q = i & 15;
      unsigned short gb = f2bf(sG[t * NT + n]);
      unsigned gpk = (unsigned)gb | ((unsigned)gb << 16);
      const unsigned* src = (const unsigned*)(sXP + (n + t) * RS) + q * 4;
      unsigned* dst = (unsigned*)(sB + n * RS) + q * 4;
#pragma unroll
      for (int e = 0; e < 4; ++e) {
        unsigned r;
        asm("v_pk_mul_bf16 %0, %1, %2" : "=v"(r) : "v"(src[e]), "v"(gpk));
        dst[e] = r;
      }
    }
    __syncthreads();
#pragma unroll
    for (int cbi = 0; cbi < 4; ++cbi) {
      const int kb = t * CC + cbi * 32;
      v16bf a = load_a_frag(Abase + kb);
#pragma unroll
      for (int nt = 0; nt < 8; ++nt) {
        v16bf bf = load_b_frag(sB + (nt * 16 + col) * RS + cbi * 32 + hi * 8);
        acc[nt] = __builtin_amdgcn_wmma_f32_16x16x32_bf16(
            false, a, false, bf, (short)0, acc[nt], false, false);
      }
    }
    __syncthreads();
  }

  // write D tile, position-major [b][n][o]
  const size_t ybase = ((size_t)b * N + n0) * CC;
  const int o0 = wave * 16 + hi * 8;
#pragma unroll
  for (int nt = 0; nt < 8; ++nt) {
    float* yp = Y + ybase + (size_t)(nt * 16 + col) * CC + o0;
    *(float4*)yp       = make_float4(acc[nt][0], acc[nt][1], acc[nt][2], acc[nt][3]);
    *(float4*)(yp + 4) = make_float4(acc[nt][4], acc[nt][5], acc[nt][6], acc[nt][7]);
  }
}

// ---------------- helper kernels ------------------------------------------
__global__ void zero_stats_kernel(float* stats) {
  stats[blockIdx.x * blockDim.x + threadIdx.x] = 0.f;
}

__global__ void wconvert_kernel(const float* __restrict__ w, unsigned short* __restrict__ Wb) {
  int o = blockIdx.x;
  for (int j = threadIdx.x; j < CC * KK; j += blockDim.x) {
    int t = j >> 7, c = j & 127;
    Wb[(size_t)o * (CC * KK) + j] = f2bf(w[((size_t)o * CC + c) * KK + t]);
  }
}

__global__ void gather_kernel(const float* __restrict__ x, const float* __restrict__ coords,
                              const int* __restrict__ idxs, unsigned short* __restrict__ XP,
                              float* __restrict__ CP, int N) {
  int bn = blockIdx.x;
  int b = bn / N, n = bn % N;
  int idx = idxs[(size_t)b * N + n];
  int c = threadIdx.x * 2;
  float v0 = x[((size_t)b * CC + c) * N + idx];
  float v1 = x[((size_t)b * CC + c + 1) * N + idx];
  *(unsigned*)&XP[(size_t)bn * CC + c] = (unsigned)f2bf(v0) | ((unsigned)f2bf(v1) << 16);
  if (threadIdx.x < 3)
    CP[((size_t)b * 3 + threadIdx.x) * N + n] = coords[((size_t)b * 3 + threadIdx.x) * N + idx];
}

__global__ void g_kernel(const float* __restrict__ CP, float* __restrict__ G, int N) {
  int b = blockIdx.y;
  int n = blockIdx.x * blockDim.x + threadIdx.x;
  if (n >= N) return;
  const float* cx = CP + (size_t)b * 3 * N;
  const float* cy = cx + N;
  const float* cz = cy + N;
  float x0 = cx[n], y0 = cy[n], z0 = cz[n];
#pragma unroll
  for (int t = 0; t < KK; ++t) {
    int p = n + t - 4;
    float g = 0.f;
    if (p >= 0 && p < N) {
      float dx = cx[p] - x0, dy = cy[p] - y0, dz = cz[p] - z0;
      g = __expf(-(dx * dx + dy * dy + dz * dz));   // SIGMA = 1
    }
    G[((size_t)b * KK + t) * N + n] = g;
  }
}

__global__ void reduce_stats_kernel(const float* __restrict__ Y, float* __restrict__ sum,
                                    float* __restrict__ sumsq, long total_pos) {
  __shared__ float ssum[256], ssq[256];
  int c = threadIdx.x & 127;
  int half = threadIdx.x >> 7;
  float s = 0.f, q = 0.f;
  for (long p = (long)blockIdx.x * 2 + half; p < total_pos; p += (long)gridDim.x * 2) {
    float v = Y[p * CC + c];
    s += v; q += v * v;
  }
  ssum[threadIdx.x] = s; ssq[threadIdx.x] = q;
  __syncthreads();
  if (threadIdx.x < 128) {
    atomicAdd(&sum[c],   ssum[c] + ssum[c + 128]);
    atomicAdd(&sumsq[c], ssq[c]  + ssq[c + 128]);
  }
}

__global__ void bn_finalize_kernel(const float* sum, const float* sumsq, const float* gamma,
                                   const float* beta, float* a, float* bb, float inv_count) {
  int c = threadIdx.x;
  float mean = sum[c] * inv_count;
  float var  = sumsq[c] * inv_count - mean * mean;
  float s = gamma[c] * rsqrtf(var + 1e-5f);
  a[c]  = s;
  bb[c] = beta[c] - mean * s;
}

__global__ void bnrelu_kernel(const float* __restrict__ Y, const float* __restrict__ a,
                              const float* __restrict__ bb, unsigned short* __restrict__ XP2,
                              long total2) {
  long i2 = (long)blockIdx.x * blockDim.x + threadIdx.x;
  if (i2 >= total2) return;
  long i = i2 * 2;
  int c = (int)(i & 127);
  float v0 = fmaxf(0.f, a[c] * Y[i] + bb[c]);
  float v1 = fmaxf(0.f, a[c + 1] * Y[i + 1] + bb[c + 1]);
  *(unsigned*)&XP2[i] = (unsigned)f2bf(v0) | ((unsigned)f2bf(v1) << 16);
}

__global__ void final_kernel(const float* __restrict__ Y2, const float* __restrict__ a,
                             const float* __restrict__ bb, const float* __restrict__ x,
                             const int* __restrict__ idxs, float* __restrict__ out, int N) {
  int bn = blockIdx.x;
  int b = bn / N, n = bn % N;
  int m = idxs[(size_t)b * N + n];
  int c = threadIdx.x;
  float y = a[c] * Y2[(size_t)bn * CC + c] + bb[c];
  size_t xoff = ((size_t)b * CC + c) * N + m;
  out[xoff] = fmaxf(0.f, y + x[xoff]);
}

// ---------------- launcher -------------------------------------------------
extern "C" void kernel_launch(void* const* d_in, const int* in_sizes, int n_in,
                              void* d_out, int out_size, void* d_ws, size_t ws_size,
                              hipStream_t stream) {
  (void)n_in; (void)out_size; (void)ws_size;
  const float* x      = (const float*)d_in[0];
  const float* coords = (const float*)d_in[1];
  const int*   indices= (const int*)d_in[2];
  // d_in[3] = reindices (unused: gather/scatter cancellation)
  const float* w1     = (const float*)d_in[4];
  const float* gamma1 = (const float*)d_in[5];
  const float* beta1  = (const float*)d_in[6];
  const float* w2     = (const float*)d_in[7];
  const float* gamma2 = (const float*)d_in[8];
  const float* beta2  = (const float*)d_in[9];
  float* out = (float*)d_out;

  const int B = 4;
  const int N = in_sizes[2] / B;

  char* ws = (char*)d_ws;
  unsigned short* XP  = (unsigned short*)ws; ws += (size_t)B * N * CC * 2;   // bf16 gathered x / bn-relu act
  float* CP = (float*)ws;                    ws += (size_t)B * 3 * N * 4;    // gathered coords
  float* G  = (float*)ws;                    ws += (size_t)B * KK * N * 4;   // gaussians
  unsigned short* W1b = (unsigned short*)ws; ws += (size_t)CC * CC * KK * 2;
  unsigned short* W2b = (unsigned short*)ws; ws += (size_t)CC * CC * KK * 2;
  float* Y = (float*)ws;                     ws += (size_t)B * N * CC * 4;   // conv output (reused)
  float* stats = (float*)ws;                                                 // 8*128 f32
  float *sum1 = stats,       *sq1 = stats + 128;
  float *sum2 = stats + 256, *sq2 = stats + 384;
  float *a1 = stats + 512, *b1 = stats + 640, *a2 = stats + 768, *b2 = stats + 896;

  const float inv_count = 1.0f / ((float)B * (float)N);
  const long total_pos = (long)B * N;
  const dim3 cgrid(N / NT, B);

  zero_stats_kernel<<<4, 256, 0, stream>>>(stats);
  wconvert_kernel<<<CC, 128, 0, stream>>>(w1, W1b);
  wconvert_kernel<<<CC, 128, 0, stream>>>(w2, W2b);
  gather_kernel<<<B * N, 64, 0, stream>>>(x, coords, indices, XP, CP, N);
  g_kernel<<<dim3((N + 255) / 256, B), 256, 0, stream>>>(CP, G, N);

  conv_wmma_kernel<<<cgrid, 256, 0, stream>>>(XP, W1b, G, Y, N);
  reduce_stats_kernel<<<512, 256, 0, stream>>>(Y, sum1, sq1, total_pos);
  bn_finalize_kernel<<<1, 128, 0, stream>>>(sum1, sq1, gamma1, beta1, a1, b1, inv_count);
  bnrelu_kernel<<<(unsigned)((total_pos * CC / 2 + 255) / 256), 256, 0, stream>>>(
      Y, a1, b1, XP, total_pos * CC / 2);

  conv_wmma_kernel<<<cgrid, 256, 0, stream>>>(XP, W2b, G, Y, N);
  reduce_stats_kernel<<<512, 256, 0, stream>>>(Y, sum2, sq2, total_pos);
  bn_finalize_kernel<<<1, 128, 0, stream>>>(sum2, sq2, gamma2, beta2, a2, b2, inv_count);
  final_kernel<<<B * N, 128, 0, stream>>>(Y, a2, b2, x, indices, out, N);
}
